// RNN_52304111731005
// MI455X (gfx1250) — matrix operand order repaired and checked by reading, not compiled, not run on hardware
//
#include <hip/hip_runtime.h>

typedef __attribute__((ext_vector_type(16))) _Float16 v16h;
typedef __attribute__((ext_vector_type(8)))  _Float16 v8h;
typedef __attribute__((ext_vector_type(4)))  _Float16 v4h;
typedef __attribute__((ext_vector_type(8)))  float    v8f;

#define WMMA_F16(a, b, c) \
  __builtin_amdgcn_wmma_f32_16x16x32_f16(false, (a), false, (b), (short)0, (c), false, false)

// Hardware transcendental tanh on CDNA5 (V_TANH_F32); libm fallback keeps it compile-safe.
__device__ inline float htanh(float x) {
#if __has_builtin(__builtin_amdgcn_tanhf)
  return __builtin_amdgcn_tanhf(x);
#elif __has_builtin(__builtin_amdgcn_tanh_f32)
  return __builtin_amdgcn_tanh_f32(x);
#else
  return tanhf(x);
#endif
}

// ---- WMMA A fragment (16x32 f16) from a row-major [16][64] f16 LDS tile ----
// lanes 0-15: M=lane,    K runs {32*ks+0..7,  32*ks+16..23}
// lanes 16-31: M=lane-16, K runs {32*ks+8..15, 32*ks+24..31}
__device__ inline v16h load_a_lds(const _Float16* h, int lane, int ks) {
  const int m  = lane & 15;
  const int hi = lane >> 4;
  const _Float16* p = h + m * 64 + ks * 32 + hi * 8;
  union { v16h v; v8h h2[2]; } u;
  u.h2[0] = *(const v8h*)(p);
  u.h2[1] = *(const v8h*)(p + 16);
  return u.v;
}

// ---- WMMA B fragment (32x16 f16) from row-major f32 weight W[N][K] ----
// B[k][n] = W[n][k]; lanes 0-15 hold K={kofs+0..7, kofs+16..23}, lanes 16-31 K={kofs+8..15, kofs+24..31}
__device__ inline v16h load_b_f32(const float* W, int ld, int n_base, int kofs, int lane) {
  const int n = n_base + (lane & 15);
  const float* p = W + n * ld + kofs + (lane >> 4) * 8;
  v16h b;
#pragma unroll
  for (int i = 0; i < 8; ++i) { b[i] = (_Float16)p[i]; b[8 + i] = (_Float16)p[16 + i]; }
  return b;
}

// ---- store 16x16 f32 D tile into row-major [16][64] f16 LDS tile at column n ----
__device__ inline void store_d_lds(_Float16* h, int n, int hi, const v8f& c) {
#pragma unroll
  for (int v = 0; v < 8; ++v) h[(v + 8 * hi) * 64 + n] = (_Float16)c[v];
}

__global__ __launch_bounds__(128)
void rnn_fused_kernel(const float* __restrict__ x,
                      const float* __restrict__ w0,   const float* __restrict__ b0,
                      const float* __restrict__ wih0, const float* __restrict__ whh0,
                      const float* __restrict__ bih0, const float* __restrict__ bhh0,
                      const float* __restrict__ wih1, const float* __restrict__ whh1,
                      const float* __restrict__ bih1, const float* __restrict__ bhh1,
                      const float* __restrict__ wih2, const float* __restrict__ whh2,
                      const float* __restrict__ bih2, const float* __restrict__ bhh2,
                      const float* __restrict__ wp1,  const float* __restrict__ bp1,
                      const float* __restrict__ wp2,  const float* __restrict__ bp2,
                      float* __restrict__ out)
{
  __shared__ __align__(16) _Float16 sS[16 * 64 * 4];    // layer-0 input seq: (row, t, feat)
  __shared__ __align__(16) _Float16 sH[3][2][16 * 64];  // per-layer ping-pong hidden tiles
  __shared__ __align__(16) float    sP1[16 * 16];

  const int tid  = threadIdx.x;
  const int lane = tid & 31;
  const int wv   = tid >> 5;        // wave 0..3 -> owns output columns [16*wv, 16*wv+16)
  const int nl   = lane & 15;
  const int hi   = lane >> 4;
  const int n    = wv * 16 + nl;    // this lane's output column in H=64
  const int row0 = blockIdx.x * 16; // batch tile base

  // ---------- weight fragments kept in registers for all 64 timesteps ----------
  const v16h Bhh0_0 = load_b_f32(whh0, 64, wv * 16,  0, lane);
  const v16h Bhh0_1 = load_b_f32(whh0, 64, wv * 16, 32, lane);
  const v16h Bih1_0 = load_b_f32(wih1, 64, wv * 16,  0, lane);
  const v16h Bih1_1 = load_b_f32(wih1, 64, wv * 16, 32, lane);
  const v16h Bhh1_0 = load_b_f32(whh1, 64, wv * 16,  0, lane);
  const v16h Bhh1_1 = load_b_f32(whh1, 64, wv * 16, 32, lane);
  const v16h Bih2_0 = load_b_f32(wih2, 64, wv * 16,  0, lane);
  const v16h Bih2_1 = load_b_f32(wih2, 64, wv * 16, 32, lane);
  const v16h Bhh2_0 = load_b_f32(whh2, 64, wv * 16,  0, lane);
  const v16h Bhh2_1 = load_b_f32(whh2, 64, wv * 16, 32, lane);

  const float wih0a = wih0[n * 4 + 0], wih0b = wih0[n * 4 + 1];
  const float wih0c = wih0[n * 4 + 2], wih0d = wih0[n * 4 + 3];
  const float bias0 = bih0[n] + bhh0[n];
  const float bias1 = bih1[n] + bhh1[n];
  const float bias2 = bih2[n] + bhh2[n];

  // ---------- fused front linear: h0 = x @ w0^T + b0, scattered as seq ----------
  // wave wv computes w0 rows [64*wv, 64*wv+64)  ==  seq feature i=wv, t = row - 64*wv
  v16h ax;
  {
    const float* px = x + (size_t)(row0 + nl) * 32 + hi * 8;
#pragma unroll
    for (int i = 0; i < 8; ++i) { ax[i] = (_Float16)px[i]; ax[8 + i] = (_Float16)px[16 + i]; }
  }
#pragma unroll
  for (int j = 0; j < 4; ++j) {
    const int ng = wv * 64 + j * 16;                  // w0 output-row block
    const v16h bw = load_b_f32(w0, 32, ng, 0, lane);  // K = 32 (single step)
    const float bb = b0[ng + nl];
    v8f c;
#pragma unroll
    for (int v = 0; v < 8; ++v) c[v] = bb;
    c = WMMA_F16(ax, bw, c);
    const int t = j * 16 + nl;
#pragma unroll
    for (int v = 0; v < 8; ++v) sS[((v + 8 * hi) * 64 + t) * 4 + wv] = (_Float16)c[v];
  }
  __syncthreads();

  // ---------- peeled t = 0 (h_{-1} == 0: no recurrent WMMAs) ----------
  {
    v8f c;
#pragma unroll
    for (int v = 0; v < 8; ++v) {
      const v4h s = *(const v4h*)&sS[((v + 8 * hi) * 64 + 0) * 4];
      c[v] = htanh(bias0 + (float)s[0] * wih0a + (float)s[1] * wih0b
                         + (float)s[2] * wih0c + (float)s[3] * wih0d);
    }
    store_d_lds(sH[0][0], n, hi, c);
    __syncthreads();

#pragma unroll
    for (int v = 0; v < 8; ++v) c[v] = bias1;
    c = WMMA_F16(load_a_lds(sH[0][0], lane, 0), Bih1_0, c);
    c = WMMA_F16(load_a_lds(sH[0][0], lane, 1), Bih1_1, c);
#pragma unroll
    for (int v = 0; v < 8; ++v) c[v] = htanh(c[v]);
    store_d_lds(sH[1][0], n, hi, c);
    __syncthreads();

#pragma unroll
    for (int v = 0; v < 8; ++v) c[v] = bias2;
    c = WMMA_F16(load_a_lds(sH[1][0], lane, 0), Bih2_0, c);
    c = WMMA_F16(load_a_lds(sH[1][0], lane, 1), Bih2_1, c);
#pragma unroll
    for (int v = 0; v < 8; ++v) c[v] = htanh(c[v]);
    store_d_lds(sH[2][0], n, hi, c);
    __syncthreads();
  }

  // ---------- steady-state: branch-free fused 3-layer recurrence ----------
  for (int t = 1; t < 64; ++t) {
    const int p = t & 1, q = p ^ 1;

    // ---- layer 0: pre (K=4, VALU) + h @ Whh0^T (WMMA) ----
    {
      v8f c;
#pragma unroll
      for (int v = 0; v < 8; ++v) {
        const v4h s = *(const v4h*)&sS[((v + 8 * hi) * 64 + t) * 4];
        c[v] = bias0 + (float)s[0] * wih0a + (float)s[1] * wih0b
                     + (float)s[2] * wih0c + (float)s[3] * wih0d;
      }
      c = WMMA_F16(load_a_lds(sH[0][q], lane, 0), Bhh0_0, c);
      c = WMMA_F16(load_a_lds(sH[0][q], lane, 1), Bhh0_1, c);
#pragma unroll
      for (int v = 0; v < 8; ++v) c[v] = htanh(c[v]);
      store_d_lds(sH[0][p], n, hi, c);
    }
    __syncthreads();

    // ---- layer 1 ----
    {
      v8f c;
#pragma unroll
      for (int v = 0; v < 8; ++v) c[v] = bias1;
      c = WMMA_F16(load_a_lds(sH[0][p], lane, 0), Bih1_0, c);
      c = WMMA_F16(load_a_lds(sH[0][p], lane, 1), Bih1_1, c);
      c = WMMA_F16(load_a_lds(sH[1][q], lane, 0), Bhh1_0, c);
      c = WMMA_F16(load_a_lds(sH[1][q], lane, 1), Bhh1_1, c);
#pragma unroll
      for (int v = 0; v < 8; ++v) c[v] = htanh(c[v]);
      store_d_lds(sH[1][p], n, hi, c);
    }
    __syncthreads();

    // ---- layer 2 ----
    {
      v8f c;
#pragma unroll
      for (int v = 0; v < 8; ++v) c[v] = bias2;
      c = WMMA_F16(load_a_lds(sH[1][p], lane, 0), Bih2_0, c);
      c = WMMA_F16(load_a_lds(sH[1][p], lane, 1), Bih2_1, c);
      c = WMMA_F16(load_a_lds(sH[2][q], lane, 0), Bhh2_0, c);
      c = WMMA_F16(load_a_lds(sH[2][q], lane, 1), Bhh2_1, c);
#pragma unroll
      for (int v = 0; v < 8; ++v) c[v] = htanh(c[v]);
      store_d_lds(sH[2][p], n, hi, c);
    }
    __syncthreads();
  }

  // ---------- predictor: relu(last @ wp1^T + bp1) @ wp2^T + bp2 (wave 0 only) ----------
  if (wv == 0) {
    const _Float16* hf = sH[2][1];  // t = 63 -> p = 1
    const float bb = bp1[nl];
    v8f c;
#pragma unroll
    for (int v = 0; v < 8; ++v) c[v] = bb;
    c = WMMA_F16(load_a_lds(hf, lane, 0), load_b_f32(wp1, 64, 0,  0, lane), c);
    c = WMMA_F16(load_a_lds(hf, lane, 1), load_b_f32(wp1, 64, 0, 32, lane), c);
#pragma unroll
    for (int v = 0; v < 8; ++v) sP1[(v + 8 * hi) * 16 + nl] = fmaxf(c[v], 0.0f);
    if (lane < 16) {
      float acc = bp2[0];
#pragma unroll
      for (int k = 0; k < 16; ++k) acc += sP1[lane * 16 + k] * wp2[k];
      out[row0 + lane] = acc;
    }
  }
}

extern "C" void kernel_launch(void* const* d_in, const int* in_sizes, int n_in,
                              void* d_out, int out_size, void* d_ws, size_t ws_size,
                              hipStream_t stream) {
  const float* x    = (const float*)d_in[0];
  const float* w0   = (const float*)d_in[1];
  const float* b0   = (const float*)d_in[2];
  const float* wih0 = (const float*)d_in[3];
  const float* whh0 = (const float*)d_in[4];
  const float* bih0 = (const float*)d_in[5];
  const float* bhh0 = (const float*)d_in[6];
  const float* wih1 = (const float*)d_in[7];
  const float* whh1 = (const float*)d_in[8];
  const float* bih1 = (const float*)d_in[9];
  const float* bhh1 = (const float*)d_in[10];
  const float* wih2 = (const float*)d_in[11];
  const float* whh2 = (const float*)d_in[12];
  const float* bih2 = (const float*)d_in[13];
  const float* bhh2 = (const float*)d_in[14];
  const float* wp1  = (const float*)d_in[15];
  const float* bp1  = (const float*)d_in[16];
  const float* wp2  = (const float*)d_in[17];
  const float* bp2  = (const float*)d_in[18];
  float* out = (float*)d_out;

  const int B = in_sizes[0] / 32;   // 16384
  dim3 grid(B / 16), block(128);
  rnn_fused_kernel<<<grid, block, 0, stream>>>(
      x, w0, b0, wih0, whh0, bih0, bhh0,
      wih1, whh1, bih1, bhh1, wih2, whh2, bih2, bhh2,
      wp1, bp1, wp2, bp2, out);
}
